// Block_3951369912372
// MI455X (gfx1250) — compile-verified, matrix operand order checked by reading, not run on hardware
//
#include <hip/hip_runtime.h>
#include <hip/hip_bf16.h>

// ---------------------------------------------------------------------------
// Types for CDNA5 WMMA (wave32): bf16 A/B fragments (16 elems/lane), f32 C/D
// ---------------------------------------------------------------------------
typedef __attribute__((ext_vector_type(16))) __bf16 v16bf;
typedef __attribute__((ext_vector_type(8)))  __bf16 v8bf;
typedef __attribute__((ext_vector_type(8)))  float  v8f;
typedef __attribute__((ext_vector_type(4)))  int    v4i;

#define DIM      768
#define HEADS    12
#define HEAD_DIM 64
#define HIDDEN   3072
#define SEQ      512
#define NROWS    16384          // 8*4*512 flattened token rows
#define SCALE    0.125f         // 64^-0.5
#define LDT      80             // padded LDS K-stride (elems) -> 160B, 16B aligned

// --- CDNA5 async global->LDS copy (ASYNCcnt-tracked), with safe fallback ---
#if defined(__has_builtin)
#  if __has_builtin(__builtin_amdgcn_global_load_async_to_lds_b128)
#    define ASYNC_LDS 1
#  endif
#endif
#ifndef ASYNC_LDS
#  define ASYNC_LDS 0
#endif

#if ASYNC_LDS
#pragma message("CDNA5: using global_load_async_to_lds_b128 staging path")
#else
#pragma message("CDNA5: async-to-LDS builtin unavailable; using VGPR staging path")
#endif

__device__ __forceinline__ void stage16B(__bf16* dst, const __bf16* src) {
#if ASYNC_LDS
    // signature (probe-confirmed): (v4i addrspace(1)*, v4i addrspace(3)*, imm offset, imm cpol)
    __builtin_amdgcn_global_load_async_to_lds_b128(
        (__attribute__((address_space(1))) v4i*)(uintptr_t)src,
        (__attribute__((address_space(3))) v4i*)(uintptr_t)dst, 0, 0);
#else
    *(v8bf*)dst = *(const v8bf*)src;     // global_load_b128 + ds_store_b128
#endif
}

__device__ __forceinline__ void async_wait() {
#if ASYNC_LDS
#  if __has_builtin(__builtin_amdgcn_s_wait_asynccnt)
    __builtin_amdgcn_s_wait_asynccnt(0);
#  else
    asm volatile("s_wait_asynccnt 0x0" ::: "memory");
#  endif
#endif
}

__device__ __forceinline__ __bf16 f2bf(float f) {
    unsigned u = __builtin_bit_cast(unsigned, f);
    u += 0x7FFFu + ((u >> 16) & 1u);          // round-to-nearest-even
    unsigned short h = (unsigned short)(u >> 16);
    return __builtin_bit_cast(__bf16, h);
}

__device__ __forceinline__ v8f vzero8() {
    v8f z;
#pragma unroll
    for (int i = 0; i < 8; ++i) z[i] = 0.0f;
    return z;
}

// Load one 16x32 bf16 A/B fragment from an LDS tile (row-major, stride LDT).
// Per-lane pattern (ISA 7.12.2): lanes 0-15 row=lane, K = {0..7,16..23};
// lanes 16-31 row=lane-16, K = {8..15,24..31}.  (A and B^T share this layout.)
// With LDT=80 both 8-elem chunks are 16B aligned -> 2x ds_load_b128.
__device__ __forceinline__ v16bf load_frag(const __bf16* tile, int rbase, int kk, int lane) {
    const __bf16* p = tile + (rbase + (lane & 15)) * LDT + kk + ((lane >> 4) << 3);
    v8bf c0 = *(const v8bf*)(p);
    v8bf c1 = *(const v8bf*)(p + 16);
    v16bf a;
#pragma unroll
    for (int e = 0; e < 8; ++e) {
        a[e]     = c0[e];
        a[8 + e] = c1[e];
    }
    return a;
}

// ---------------------------------------------------------------------------
// f32 -> bf16 conversion (weights / one-shot)
// ---------------------------------------------------------------------------
__global__ __launch_bounds__(256) void cvt_kernel(const float* __restrict__ in,
                                                  __bf16* __restrict__ out, int n) {
    for (int i = blockIdx.x * 256 + threadIdx.x; i < n; i += gridDim.x * 256)
        out[i] = f2bf(in[i]);
}

// ---------------------------------------------------------------------------
// LayerNorm over D=768, one block (256 thr) per row, bf16 output
// ---------------------------------------------------------------------------
__global__ __launch_bounds__(256) void ln_kernel(const float* __restrict__ x,
                                                 const float* __restrict__ g,
                                                 const float* __restrict__ b,
                                                 __bf16* __restrict__ out) {
    __shared__ float rs[8], rs2[8];
    const int tid = threadIdx.x, lane = tid & 31, wave = tid >> 5;
    const float* xr = x + (size_t)blockIdx.x * DIM;
    float v0 = xr[tid], v1 = xr[tid + 256], v2 = xr[tid + 512];
    float s = v0 + v1 + v2;
    float s2 = v0 * v0 + v1 * v1 + v2 * v2;
#pragma unroll
    for (int off = 16; off > 0; off >>= 1) {
        s  += __shfl_down(s,  off, 32);
        s2 += __shfl_down(s2, off, 32);
    }
    if (lane == 0) { rs[wave] = s; rs2[wave] = s2; }
    __syncthreads();
    float ts = 0.f, ts2 = 0.f;
#pragma unroll
    for (int w = 0; w < 8; ++w) { ts += rs[w]; ts2 += rs2[w]; }
    const float mean = ts * (1.0f / DIM);
    const float var  = ts2 * (1.0f / DIM) - mean * mean;
    const float inv  = rsqrtf(var + 1e-5f);
    __bf16* o = out + (size_t)blockIdx.x * DIM;
    o[tid]       = f2bf((v0 - mean) * inv * g[tid]       + b[tid]);
    o[tid + 256] = f2bf((v1 - mean) * inv * g[tid + 256] + b[tid + 256]);
    o[tid + 512] = f2bf((v2 - mean) * inv * g[tid + 512] + b[tid + 512]);
}

// ---------------------------------------------------------------------------
// bf16 WMMA GEMM: C[M,N] = A[M,K] @ B[K,N]  (+bias) (+GELU) (+resid, f32 out)
// Block = 256 thr (8 waves), tile 128x128, K-step 64. Wave tile 32x64.
// A-tile staged via async global->LDS; B-tile transposed via VGPR scatter.
// ---------------------------------------------------------------------------
template <bool BIAS, bool GELU, bool BF16OUT>
__global__ __launch_bounds__(256) void gemm_bf16(const __bf16* __restrict__ A,
                                                 const __bf16* __restrict__ B,
                                                 const float* __restrict__ bias,
                                                 const float* __restrict__ resid,
                                                 void* __restrict__ outp,
                                                 int M, int N, int K) {
    __shared__ __bf16 sA[128 * LDT];
    __shared__ __bf16 sB[128 * LDT];   // holds B^T tile: [n][k]

    const int tid = threadIdx.x, lane = tid & 31, wave = tid >> 5;
    const int m0 = blockIdx.y * 128, n0 = blockIdx.x * 128;
    const int wm = (wave & 3) * 32;    // wave M offset in tile
    const int wn = (wave >> 2) * 64;   // wave N offset in tile

    v8f acc[2][4];
#pragma unroll
    for (int mi = 0; mi < 2; ++mi)
#pragma unroll
        for (int ni = 0; ni < 4; ++ni) acc[mi][ni] = vzero8();

    for (int k0 = 0; k0 < K; k0 += 64) {
        // stage A 128x64 (row-major) -- async DMA into LDS
#pragma unroll
        for (int i = 0; i < 4; ++i) {
            int c = tid + i * 256;               // 1024 chunks of 8 elems
            int row = c >> 3, kc = (c & 7) * 8;
            stage16B(sA + row * LDT + kc, A + (size_t)(m0 + row) * K + k0 + kc);
        }
        // stage B^T 128x64 (global B is [K][N] row-major) -- transpose scatter
#pragma unroll
        for (int i = 0; i < 4; ++i) {
            int c = tid + i * 256;               // 64(k) x 16 chunks(n)
            int kr = c >> 4, nc = (c & 15) * 8;
            v8bf v = *(const v8bf*)(B + (size_t)(k0 + kr) * N + n0 + nc);
#pragma unroll
            for (int e = 0; e < 8; ++e) sB[(nc + e) * LDT + kr] = v[e];
        }
        if (k0 + 64 < K) {
            __builtin_prefetch(A + (size_t)(m0 + (tid >> 1)) * K + k0 + 64, 0, 1);
            __builtin_prefetch(B + (size_t)(k0 + 64 + (tid >> 2)) * N + n0, 0, 1);
        }
        async_wait();
        __syncthreads();

#pragma unroll
        for (int kk = 0; kk < 64; kk += 32) {
            v16bf af[2], bfr[4];
#pragma unroll
            for (int mi = 0; mi < 2; ++mi) af[mi] = load_frag(sA, wm + 16 * mi, kk, lane);
#pragma unroll
            for (int ni = 0; ni < 4; ++ni) bfr[ni] = load_frag(sB, wn + 16 * ni, kk, lane);
#pragma unroll
            for (int mi = 0; mi < 2; ++mi)
#pragma unroll
                for (int ni = 0; ni < 4; ++ni)
                    acc[mi][ni] = __builtin_amdgcn_wmma_f32_16x16x32_bf16(
                        false, af[mi], false, bfr[ni], (short)0, acc[mi][ni], false, false);
        }
        __syncthreads();
    }

    // epilogue (C layout: VGPR r -> row r / r+8; col = lane&15 within 16-tile)
    const int col = lane & 15, half = lane >> 4;
#pragma unroll
    for (int mi = 0; mi < 2; ++mi)
#pragma unroll
        for (int ni = 0; ni < 4; ++ni) {
            const int gcol = n0 + wn + ni * 16 + col;
            const float bv = BIAS ? bias[gcol] : 0.0f;
#pragma unroll
            for (int r = 0; r < 8; ++r) {
                const int grow = m0 + wm + mi * 16 + half * 8 + r;
                float v = acc[mi][ni][r] + bv;
                if constexpr (GELU)
                    v = 0.5f * v * (1.0f + erff(v * 0.70710678118f));
                const size_t idx = (size_t)grow * N + gcol;
                if constexpr (BF16OUT) {
                    ((__bf16*)outp)[idx] = f2bf(v);
                } else {
                    ((float*)outp)[idx] = v + resid[idx];
                }
            }
        }
}

// ---------------------------------------------------------------------------
// Flash attention, bf16 WMMA, f32 accumulation.
// Block = 128 thr (4 waves); each wave owns 16 query rows; block owns 64 rows.
// grid = (8 row-blocks, 12 heads, 32 (b,s)).  qkv row = [q|k|v] x heads x 64.
// Q/K staged via async global->LDS; V staged transposed via VGPR scatter.
// ---------------------------------------------------------------------------
__global__ __launch_bounds__(128) void attn_kernel(const __bf16* __restrict__ qkv,
                                                   __bf16* __restrict__ y) {
    __shared__ __bf16 sQ[64 * LDT];
    __shared__ __bf16 sK[64 * LDT];
    __shared__ __bf16 sVt[64 * LDT];        // V transposed: [d][token]
    __shared__ __bf16 sP[4][16 * LDT];      // wave-private probability tiles

    const int tid = threadIdx.x, lane = tid & 31, wave = tid >> 5;
    const int col = lane & 15, half = lane >> 4;
    const int rb = blockIdx.x, head = blockIdx.y, bs = blockIdx.z;

    const size_t rstride = 3 * DIM;         // 2304 bf16 per token row
    const __bf16* qbase = qkv + (size_t)bs * SEQ * rstride + head * HEAD_DIM;
    const __bf16* kbase = qbase + DIM;
    const __bf16* vbase = qbase + 2 * DIM;

    // stage Q block (64 rows x 64 dims) -- async
#pragma unroll
    for (int i = 0; i < 4; ++i) {
        int c = tid + i * 128;              // 512 chunks of 8
        int row = c >> 3, kc = (c & 7) * 8;
        stage16B(sQ + row * LDT + kc, qbase + (size_t)(rb * 64 + row) * rstride + kc);
    }

    v8f acc[4];
    float mprev[8], lsum[8];
#pragma unroll
    for (int ni = 0; ni < 4; ++ni) acc[ni] = vzero8();
#pragma unroll
    for (int r = 0; r < 8; ++r) { mprev[r] = -3.0e38f; lsum[r] = 0.0f; }

    for (int j = 0; j < SEQ / 64; ++j) {
        __syncthreads();                    // protect sK/sVt reuse
        // stage K block (row-major: B^T layout for S = Q K^T) -- async
#pragma unroll
        for (int i = 0; i < 4; ++i) {
            int c = tid + i * 128;
            int row = c >> 3, kc = (c & 7) * 8;
            stage16B(sK + row * LDT + kc, kbase + (size_t)(j * 64 + row) * rstride + kc);
        }
        // stage V transposed -- VGPR scatter
#pragma unroll
        for (int i = 0; i < 4; ++i) {
            int c = tid + i * 128;
            int row = c >> 3, kc = (c & 7) * 8;
            v8bf v = *(const v8bf*)(vbase + (size_t)(j * 64 + row) * rstride + kc);
#pragma unroll
            for (int e = 0; e < 8; ++e) sVt[(kc + e) * LDT + row] = v[e];
        }
        async_wait();                       // covers this iter's K and initial Q
        __syncthreads();

        // S = Q @ K^T
        v8f s[4];
#pragma unroll
        for (int ni = 0; ni < 4; ++ni) s[ni] = vzero8();
#pragma unroll
        for (int kk = 0; kk < 64; kk += 32) {
            v16bf aq = load_frag(sQ, wave * 16, kk, lane);
#pragma unroll
            for (int ni = 0; ni < 4; ++ni) {
                v16bf bk = load_frag(sK, ni * 16, kk, lane);
                s[ni] = __builtin_amdgcn_wmma_f32_16x16x32_bf16(
                    false, aq, false, bk, (short)0, s[ni], false, false);
            }
        }

        // online softmax (rows live across the 16 lanes of each half-wave)
#pragma unroll
        for (int r = 0; r < 8; ++r) {
            float mx = -3.0e38f;
#pragma unroll
            for (int ni = 0; ni < 4; ++ni) {
                s[ni][r] *= SCALE;
                mx = fmaxf(mx, s[ni][r]);
            }
#pragma unroll
            for (int off = 8; off > 0; off >>= 1)
                mx = fmaxf(mx, __shfl_xor(mx, off, 32));
            const float mnew = fmaxf(mprev[r], mx);
            const float corr = __expf(mprev[r] - mnew);
            float ps = 0.0f;
#pragma unroll
            for (int ni = 0; ni < 4; ++ni) {
                float p = __expf(s[ni][r] - mnew);
                s[ni][r] = p;
                ps += p;
            }
#pragma unroll
            for (int off = 8; off > 0; off >>= 1)
                ps += __shfl_xor(ps, off, 32);
            lsum[r] = lsum[r] * corr + ps;
#pragma unroll
            for (int ni = 0; ni < 4; ++ni) acc[ni][r] *= corr;
            mprev[r] = mnew;
        }

        // P (C layout) -> LDS -> A-fragment layout
#pragma unroll
        for (int r = 0; r < 8; ++r)
#pragma unroll
            for (int ni = 0; ni < 4; ++ni)
                sP[wave][(half * 8 + r) * LDT + ni * 16 + col] = f2bf(s[ni][r]);
        asm volatile("s_wait_dscnt 0x0" ::: "memory");   // same-wave LDS RAW

        // O += P @ V
#pragma unroll
        for (int kk = 0; kk < 64; kk += 32) {
            v16bf ap = load_frag(&sP[wave][0], 0, kk, lane);
#pragma unroll
            for (int ni = 0; ni < 4; ++ni) {
                v16bf bv = load_frag(sVt, ni * 16, kk, lane);
                acc[ni] = __builtin_amdgcn_wmma_f32_16x16x32_bf16(
                    false, ap, false, bv, (short)0, acc[ni], false, false);
            }
        }
    }

    // normalize and store y (bf16, [token][dim] with head offset)
    __bf16* ybase = y + (size_t)bs * SEQ * DIM + head * HEAD_DIM;
#pragma unroll
    for (int ni = 0; ni < 4; ++ni)
#pragma unroll
        for (int r = 0; r < 8; ++r) {
            const int row = rb * 64 + wave * 16 + half * 8 + r;
            ybase[(size_t)row * DIM + ni * 16 + col] = f2bf(acc[ni][r] / lsum[r]);
        }
}

// ---------------------------------------------------------------------------
// Host-side orchestration
// ---------------------------------------------------------------------------
extern "C" void kernel_launch(void* const* d_in, const int* in_sizes, int n_in,
                              void* d_out, int out_size, void* d_ws, size_t ws_size,
                              hipStream_t stream) {
    (void)in_sizes; (void)n_in; (void)out_size; (void)ws_size;
    const float* x    = (const float*)d_in[0];
    const float* g1   = (const float*)d_in[1];
    const float* b1   = (const float*)d_in[2];
    const float* Wqkv = (const float*)d_in[3];
    const float* Wp   = (const float*)d_in[4];
    const float* bp   = (const float*)d_in[5];
    const float* g2   = (const float*)d_in[6];
    const float* b2   = (const float*)d_in[7];
    const float* W1   = (const float*)d_in[8];
    const float* bm1  = (const float*)d_in[9];
    const float* W2   = (const float*)d_in[10];
    const float* bm2  = (const float*)d_in[11];
    float* out = (float*)d_out;

    char* ws = (char*)d_ws;
    __bf16* wqkv = (__bf16*)(ws + 0);                       // 768*2304
    __bf16* wp   = (__bf16*)(ws + 3538944);                 // 768*768
    __bf16* w1   = (__bf16*)(ws + 4718592);                 // 768*3072
    __bf16* w2   = (__bf16*)(ws + 9437184);                 // 3072*768
    __bf16* h    = (__bf16*)(ws + 14155776);                // 16384*768 (LN1, then LN2)
    __bf16* qkvb = (__bf16*)(ws + 39321600);                // 16384*2304
    __bf16* yb   = (__bf16*)(ws + 114819072);               // 16384*768
    __bf16* mb   = (__bf16*)(ws + 39321600);                // 16384*3072, aliases qkv+y

    auto cvt = [&](const float* src, __bf16* dst, int n) {
        cvt_kernel<<<(n + 255) / 256, 256, 0, stream>>>(src, dst, n);
    };
    cvt(Wqkv, wqkv, DIM * 3 * DIM);
    cvt(Wp,   wp,   DIM * DIM);
    cvt(W1,   w1,   DIM * HIDDEN);
    cvt(W2,   w2,   HIDDEN * DIM);

    // 1) h = LN1(x) in bf16
    ln_kernel<<<NROWS, 256, 0, stream>>>(x, g1, b1, h);

    // 2) qkv = h @ Wqkv   (bf16 out)
    gemm_bf16<false, false, true><<<dim3((3 * DIM) / 128, NROWS / 128), 256, 0, stream>>>(
        h, wqkv, nullptr, nullptr, qkvb, NROWS, 3 * DIM, DIM);

    // 3) y = attention(qkv)  (bf16)
    attn_kernel<<<dim3(SEQ / 64, HEADS, 32), 128, 0, stream>>>(qkvb, yb);

    // 4) x2 = x + y @ Wp + bp  (f32, into d_out)
    gemm_bf16<true, false, false><<<dim3(DIM / 128, NROWS / 128), 256, 0, stream>>>(
        yb, wp, bp, x, out, NROWS, DIM, DIM);

    // 5) h = LN2(x2) in bf16
    ln_kernel<<<NROWS, 256, 0, stream>>>(out, g2, b2, h);

    // 6) m = gelu(h @ W1 + bm1)  (bf16)
    gemm_bf16<true, true, true><<<dim3(HIDDEN / 128, NROWS / 128), 256, 0, stream>>>(
        h, w1, bm1, nullptr, mb, NROWS, HIDDEN, DIM);

    // 7) out = x2 + m @ W2 + bm2  (f32, residual read from d_out itself)
    gemm_bf16<true, false, false><<<dim3(DIM / 128, NROWS / 128), 256, 0, stream>>>(
        mb, w2, bm2, out, out, NROWS, DIM, HIDDEN);
}
